// RWKV_TimeMix_30923764532055
// MI455X (gfx1250) — compile-verified, matrix operand order checked by reading, not run on hardware
//
#include <hip/hip_runtime.h>
#include <hip/hip_bf16.h>
#include <math.h>

// ---------------------------------------------------------------------------
// RWKV-6 TimeMix forward for MI455X (gfx1250, wave32, WMMA bf16)
// B=2, T=2048, C=1024, H=16, N=64
// ---------------------------------------------------------------------------

typedef __attribute__((ext_vector_type(16))) __bf16 bf16x16;
typedef __attribute__((ext_vector_type(8)))  __bf16 bf16x8;
typedef __attribute__((ext_vector_type(4)))  __bf16 bf16x4;
typedef __attribute__((ext_vector_type(8)))  float  f32x8;
typedef __attribute__((ext_vector_type(4)))  float  f32x4;

#define RWKV_C   1024
#define RWKV_T   2048
#define RWKV_H   16
#define RWKV_N   64
#define RWKV_ROWS 4096          // B*T
#define GK_LDA   40             // padded LDS stride (elements) for GEMM tiles

// CDNA5 async global->LDS path (ASYNCcnt-tracked), with safe fallback.
#if defined(__AMDGCN__) && __has_builtin(__builtin_amdgcn_global_load_async_to_lds_b128) && __has_builtin(__builtin_amdgcn_s_wait_asynccnt)
#define USE_ASYNC_LDS 1
#else
#define USE_ASYNC_LDS 0
#endif

#if USE_ASYNC_LDS
// Builtin signature (from hipcc diagnostic): param1 = addrspace(1) ptr to
// 'int __vector(4)' (b128 payload), param2 = LDS side, then (offset, cpol).
typedef int v4i_vs __attribute__((vector_size(16)));
typedef __attribute__((address_space(1))) v4i_vs* async_gptr_t;
typedef __attribute__((address_space(3))) v4i_vs* async_lptr_t;

__device__ __forceinline__ void async_copy_b128(const __bf16* g, __bf16* l) {
  __builtin_amdgcn_global_load_async_to_lds_b128(
      (async_gptr_t)g, (async_lptr_t)l, 0, 0);
}
#endif

// Load one 16x32 bf16 WMMA operand fragment from an LDS tile.
// A/B 16-bit layout: lane<16 -> row=lane, K = 0..7 & 16..23 ; lane>=16 -> K = 8..15 & 24..31
__device__ __forceinline__ bf16x16 ld_frag(const __bf16* bp, int row0, int stride,
                                           int k0, int frow, int fk) {
  const __bf16* p = bp + (row0 + frow) * stride + k0 + fk;
  const bf16x8 lo = *(const bf16x8*)p;
  const bf16x8 hi = *(const bf16x8*)(p + 16);
  bf16x16 f;
#pragma unroll
  for (int i = 0; i < 8; ++i) { f[i] = lo[i]; f[i + 8] = hi[i]; }
  return f;
}

// ---------------------------------------------------------------------------
// K0: fp32 [K,N] weight -> bf16 [N,K] (transpose + convert)
// ---------------------------------------------------------------------------
__global__ __launch_bounds__(256) void wconv_kernel(const float* __restrict__ W,
                                                    __bf16* __restrict__ WT) {
  __shared__ float tile[32][33];
  const int tx = threadIdx.x & 31;
  const int ty = threadIdx.x >> 5;            // 0..7
  const int bx = blockIdx.x;                  // n tile
  const int by = blockIdx.y;                  // k tile
#pragma unroll
  for (int p = 0; p < 4; ++p)
    tile[ty + p * 8][tx] =
        W[(size_t)(by * 32 + ty + p * 8) * RWKV_C + bx * 32 + tx];
  __syncthreads();
#pragma unroll
  for (int p = 0; p < 4; ++p)
    WT[(size_t)(bx * 32 + ty + p * 8) * RWKV_C + by * 32 + tx] =
        (__bf16)tile[tx][ty + p * 8];
}

// ---------------------------------------------------------------------------
// K1: token shift -> xxx = x + (x_prev - x)*tmx ; t5 = tanh(xxx @ maa_w1)
// ---------------------------------------------------------------------------
__global__ __launch_bounds__(256) void mix1_kernel(const float* __restrict__ x,
                                                   const float* __restrict__ tmx,
                                                   const float* __restrict__ w1,  // [1024,160]
                                                   float* __restrict__ t5) {
  const int row = blockIdx.x;
  const int t   = row & (RWKV_T - 1);
  const int tid = threadIdx.x;
  __shared__ float sxxx[RWKV_C];
  const float* xr = x + (size_t)row * RWKV_C;
  const float* xp = xr - RWKV_C;
#pragma unroll
  for (int i = 0; i < 4; ++i) {
    const int c = tid * 4 + i;
    const float xv = xr[c];
    float pv = 0.0f;
    if (t != 0) pv = xp[c];
    sxxx[c] = xv + (pv - xv) * tmx[c];
  }
  __syncthreads();
  if (tid < 160) {
    float acc = 0.0f;
    for (int k = 0; k < RWKV_C; ++k) acc += sxxx[k] * w1[k * 160 + tid];
    t5[(size_t)row * 160 + tid] = tanhf(acc);
  }
}

// ---------------------------------------------------------------------------
// K2: m[f] = t5[f] @ maa_w2[f]; produce xw (f32) and xk,xv,xr,xg (bf16)
// ---------------------------------------------------------------------------
__global__ __launch_bounds__(256) void mix2_kernel(
    const float* __restrict__ x,   const float* __restrict__ tmx,
    const float* __restrict__ tmw, const float* __restrict__ tmk,
    const float* __restrict__ tmv, const float* __restrict__ tmr,
    const float* __restrict__ tmg,
    const float* __restrict__ w2,  // [5,32,1024]
    const float* __restrict__ t5,  // [rows,160]
    float*  __restrict__ xw,
    __bf16* __restrict__ xk, __bf16* __restrict__ xv,
    __bf16* __restrict__ xrb, __bf16* __restrict__ xg) {
  const int row = blockIdx.x;
  const int t   = row & (RWKV_T - 1);
  const int tid = threadIdx.x;
  __shared__ float st5[160];
  if (tid < 160) st5[tid] = t5[(size_t)row * 160 + tid];
  __syncthreads();

  const int c0 = tid * 4;
  const float* xr = x + (size_t)row * RWKV_C;
  const float* xp = xr - RWKV_C;

  f32x4 xv4 = *(const f32x4*)(xr + c0);
  f32x4 pv4 = (f32x4)0.0f;
  if (t != 0) pv4 = *(const f32x4*)(xp + c0);
  f32x4 xx4;
#pragma unroll
  for (int i = 0; i < 4; ++i) xx4[i] = pv4[i] - xv4[i];

  f32x4 m[5];
#pragma unroll
  for (int f = 0; f < 5; ++f) {
    f32x4 acc = (f32x4)0.0f;
    for (int e = 0; e < 32; ++e) {
      const float tv = st5[f * 32 + e];
      const f32x4 wv = *(const f32x4*)(w2 + (size_t)(f * 32 + e) * RWKV_C + c0);
#pragma unroll
      for (int i = 0; i < 4; ++i) acc[i] += tv * wv[i];
    }
    m[f] = acc;
  }

  const f32x4 vw = *(const f32x4*)(tmw + c0);
  const f32x4 vk = *(const f32x4*)(tmk + c0);
  const f32x4 vv = *(const f32x4*)(tmv + c0);
  const f32x4 vr = *(const f32x4*)(tmr + c0);
  const f32x4 vg = *(const f32x4*)(tmg + c0);

  f32x4 ow;
  bf16x4 ok, ov, orr, og;
#pragma unroll
  for (int i = 0; i < 4; ++i) {
    ow[i]  =          xv4[i] + xx4[i] * (vw[i] + m[0][i]);
    ok[i]  = (__bf16)(xv4[i] + xx4[i] * (vk[i] + m[1][i]));
    ov[i]  = (__bf16)(xv4[i] + xx4[i] * (vv[i] + m[2][i]));
    orr[i] = (__bf16)(xv4[i] + xx4[i] * (vr[i] + m[3][i]));
    og[i]  = (__bf16)(xv4[i] + xx4[i] * (vg[i] + m[4][i]));
  }
  const size_t o = (size_t)row * RWKV_C + c0;
  *(f32x4*)(xw + o)   = ow;
  *(bf16x4*)(xk + o)  = ok;
  *(bf16x4*)(xv + o)  = ov;
  *(bf16x4*)(xrb + o) = orr;
  *(bf16x4*)(xg + o)  = og;
}

// ---------------------------------------------------------------------------
// K3: decay = exp(-exp(tdecay + tanh(xw @ td_w1) @ td_w2))
// ---------------------------------------------------------------------------
__global__ __launch_bounds__(256) void decay_kernel(const float* __restrict__ xw,
                                                    const float* __restrict__ td,
                                                    const float* __restrict__ w1, // [1024,64]
                                                    const float* __restrict__ w2, // [64,1024]
                                                    float* __restrict__ decay) {
  const int row = blockIdx.x;
  const int tid = threadIdx.x;
  __shared__ float sx[RWKV_C];
  __shared__ float sh[64];
  const float* xr = xw + (size_t)row * RWKV_C;
#pragma unroll
  for (int i = 0; i < 4; ++i) sx[tid * 4 + i] = xr[tid * 4 + i];
  __syncthreads();
  if (tid < 64) {
    float acc = 0.0f;
    for (int k = 0; k < RWKV_C; ++k) acc += sx[k] * w1[k * 64 + tid];
    sh[tid] = tanhf(acc);
  }
  __syncthreads();
  const int c0 = tid * 4;
  f32x4 acc = *(const f32x4*)(td + c0);
  for (int e = 0; e < 64; ++e) {
    const float hv = sh[e];
    const f32x4 wv = *(const f32x4*)(w2 + (size_t)e * RWKV_C + c0);
#pragma unroll
    for (int i = 0; i < 4; ++i) acc[i] += hv * wv[i];
  }
#pragma unroll
  for (int i = 0; i < 4; ++i)
    decay[(size_t)row * RWKV_C + c0 + i] = __expf(-__expf(acc[i]));
}

// ---------------------------------------------------------------------------
// K4: bf16 WMMA GEMM  C[M,N] = A[M,K] @ BT[N,K]^T   (f32 accumulate)
// block tile 128x128, 8 waves; wave tile 32(M) x 64(N) = 2x4 WMMA tiles.
// Double-buffered LDS; tiles moved with async global->LDS when available.
// act: 0 = none, 1 = silu
// ---------------------------------------------------------------------------
__global__ __launch_bounds__(256) void gemm_bf16_kernel(
    const __bf16* __restrict__ A, const __bf16* __restrict__ BT,
    float* __restrict__ C, int M, int N, int K, int act) {
  __shared__ __bf16 As[2][128 * GK_LDA];
  __shared__ __bf16 Bs[2][128 * GK_LDA];

  const int tid  = threadIdx.x;
  const int lane = tid & 31;
  const int wave = tid >> 5;       // 0..7
  const int wm   = wave >> 1;      // 0..3 -> M offset wm*32
  const int wn   = wave & 1;       // 0..1 -> N offset wn*64
  const int bm   = blockIdx.y * 128;
  const int bn   = blockIdx.x * 128;

  f32x8 acc[2][4];
#pragma unroll
  for (int mt = 0; mt < 2; ++mt)
#pragma unroll
    for (int nt = 0; nt < 4; ++nt)
#pragma unroll
      for (int p = 0; p < 8; ++p) acc[mt][nt][p] = 0.0f;

  const int lrow = tid >> 2;          // 0..63
  const int lcol = (tid & 3) * 8;     // 0,8,16,24
  const int frow = lane & 15;
  const int fk   = (lane >> 4) * 8;

  const int la0 = lrow * GK_LDA + lcol;             // LDS dest, rows 0..63
  const int la1 = (lrow + 64) * GK_LDA + lcol;      // LDS dest, rows 64..127
  const __bf16* gA0 = A  + (size_t)(bm + lrow) * K + lcol;
  const __bf16* gA1 = A  + (size_t)(bm + lrow + 64) * K + lcol;
  const __bf16* gB0 = BT + (size_t)(bn + lrow) * K + lcol;
  const __bf16* gB1 = BT + (size_t)(bn + lrow + 64) * K + lcol;

  const int NT = K / 32;

  // ---- prologue: fetch tile 0 into buffer 0 ----
#if USE_ASYNC_LDS
  async_copy_b128(gA0, &As[0][la0]);
  async_copy_b128(gA1, &As[0][la1]);
  async_copy_b128(gB0, &Bs[0][la0]);
  async_copy_b128(gB1, &Bs[0][la1]);
#else
  *(bf16x8*)&As[0][la0] = *(const bf16x8*)gA0;
  *(bf16x8*)&As[0][la1] = *(const bf16x8*)gA1;
  *(bf16x8*)&Bs[0][la0] = *(const bf16x8*)gB0;
  *(bf16x8*)&Bs[0][la1] = *(const bf16x8*)gB1;
#endif

  for (int it = 0; it < NT; ++it) {
    const int cur = it & 1;
    const int nxt = cur ^ 1;

#if USE_ASYNC_LDS
    __builtin_amdgcn_s_wait_asynccnt(0);
    __syncthreads();
    if (it + 1 < NT) {
      const int ko = (it + 1) * 32;
      async_copy_b128(gA0 + ko, &As[nxt][la0]);
      async_copy_b128(gA1 + ko, &As[nxt][la1]);
      async_copy_b128(gB0 + ko, &Bs[nxt][la0]);
      async_copy_b128(gB1 + ko, &Bs[nxt][la1]);
    }
#else
    __syncthreads();
    bf16x8 na0 = {}, na1 = {}, nb0 = {}, nb1 = {};
    if (it + 1 < NT) {
      const int ko = (it + 1) * 32;
      na0 = *(const bf16x8*)(gA0 + ko);
      na1 = *(const bf16x8*)(gA1 + ko);
      nb0 = *(const bf16x8*)(gB0 + ko);
      nb1 = *(const bf16x8*)(gB1 + ko);
    }
#endif

    // ---- compute on current buffer ----
    bf16x16 afrag[2];
#pragma unroll
    for (int mt = 0; mt < 2; ++mt)
      afrag[mt] = ld_frag(As[cur], wm * 32 + mt * 16, GK_LDA, 0, frow, fk);
#pragma unroll
    for (int nt = 0; nt < 4; ++nt) {
      const bf16x16 bfrag = ld_frag(Bs[cur], wn * 64 + nt * 16, GK_LDA, 0, frow, fk);
#pragma unroll
      for (int mt = 0; mt < 2; ++mt) {
        acc[mt][nt] = __builtin_amdgcn_wmma_f32_16x16x32_bf16(
            false, afrag[mt], false, bfrag, (short)0, acc[mt][nt], false, false);
      }
    }

#if !USE_ASYNC_LDS
    if (it + 1 < NT) {
      *(bf16x8*)&As[nxt][la0] = na0;
      *(bf16x8*)&As[nxt][la1] = na1;
      *(bf16x8*)&Bs[nxt][la0] = nb0;
      *(bf16x8*)&Bs[nxt][la1] = nb1;
    }
#endif
  }

  const int rowoff = (lane >> 4) * 8;   // D layout: lanes 16..31 hold M = p+8
  const int colo   = lane & 15;
#pragma unroll
  for (int mt = 0; mt < 2; ++mt) {
#pragma unroll
    for (int nt = 0; nt < 4; ++nt) {
      const int col   = bn + wn * 64 + nt * 16 + colo;
      const int rbase = bm + wm * 32 + mt * 16 + rowoff;
#pragma unroll
      for (int p = 0; p < 8; ++p) {
        float v = acc[mt][nt][p];
        if (act == 1) v = v / (1.0f + __expf(-v));   // silu
        C[(size_t)(rbase + p) * N + col] = v;
      }
    }
  }
}

// ---------------------------------------------------------------------------
// K5: chunked WKV recurrence via WMMA. one block per (b,h); 256 threads.
// chunk L=32.  A_t = cumprod(decay), r~_t = r_t*A_{t-1}, k~_t = k_t/A_t.
//   P[t][s] = r~_t . k~_s   (s<t) ; P[t][t] = sum_i r u k ; else 0
//   y = P @ V + r~ @ S0
//   S  = diag(A_L) S + (A_L*k~)^T @ V
// State kept transposed in LDS: S0T[j][i] (f32) + bf16 copy for the B operand.
// ---------------------------------------------------------------------------
#define WKV_L 32
__global__ __launch_bounds__(256) void wkv_chunk_kernel(
    const float* __restrict__ r, const float* __restrict__ k,
    const float* __restrict__ v, const float* __restrict__ d,
    const float* __restrict__ u,   // [H,N]
    float* __restrict__ y) {
  const int bh   = blockIdx.x;   // 0..31
  const int b    = bh >> 4;
  const int h    = bh & 15;
  const int tid  = threadIdx.x;
  const int lane = tid & 31;
  const int wave = tid >> 5;

  __shared__ float  Rc[32 * 68];     // r chunk  [t][i]
  __shared__ float  Kc[32 * 68];     // k chunk
  __shared__ float  Ac[32 * 68];     // decay -> cumprod in place
  __shared__ float  S0T[64 * 68];    // state^T [j][i] f32
  __shared__ __bf16 S0b[64 * 72];    // bf16 copy of S0T (B operand, rows=j, K=i)
  __shared__ __bf16 Rt[32 * 72];     // r~ bf16 [t][i]   (A operand)
  __shared__ __bf16 Kt[32 * 72];     // k~ bf16 [t][i]   (B operand rows=s, K=i)
  __shared__ __bf16 BkT[64 * 40];    // (aL*k~)^T [i][s] (A operand)
  __shared__ __bf16 Vt[64 * 40];     // v^T [j][s]       (B operand rows=j, K=s)
  __shared__ __bf16 Pm[32 * 40];     // masked P [t][s]  (A operand)
  __shared__ float  aL[64];
  __shared__ float  sdiag[32];
  __shared__ float  su[64];

  for (int idx = tid; idx < 64 * 68; idx += 256) S0T[idx] = 0.0f;
  if (tid < 64) su[tid] = u[h * 64 + tid];
  __syncthreads();

  const size_t base = (size_t)b * RWKV_T * RWKV_C + h * 64;
  const int lt = tid >> 3;           // 0..31 local time step
  const int li = (tid & 7) * 8;      // channel start
  const int frow = lane & 15;
  const int fk   = (lane >> 4) * 8;

  for (int c = 0; c < RWKV_T / WKV_L; ++c) {
    const int t0 = c * WKV_L;
    // ---- load chunk, transpose V ----
    {
      const size_t off = base + (size_t)(t0 + lt) * RWKV_C + li;
      const f32x4 r0 = *(const f32x4*)(r + off), r1 = *(const f32x4*)(r + off + 4);
      const f32x4 k0 = *(const f32x4*)(k + off), k1 = *(const f32x4*)(k + off + 4);
      const f32x4 d0 = *(const f32x4*)(d + off), d1 = *(const f32x4*)(d + off + 4);
      const f32x4 v0 = *(const f32x4*)(v + off), v1 = *(const f32x4*)(v + off + 4);
      *(f32x4*)&Rc[lt * 68 + li] = r0; *(f32x4*)&Rc[lt * 68 + li + 4] = r1;
      *(f32x4*)&Kc[lt * 68 + li] = k0; *(f32x4*)&Kc[lt * 68 + li + 4] = k1;
      *(f32x4*)&Ac[lt * 68 + li] = d0; *(f32x4*)&Ac[lt * 68 + li + 4] = d1;
#pragma unroll
      for (int q = 0; q < 4; ++q) {
        Vt[(li + q) * 40 + lt]     = (__bf16)v0[q];
        Vt[(li + 4 + q) * 40 + lt] = (__bf16)v1[q];
      }
    }
    __syncthreads();
    // ---- per-channel cumulative decay product ----
    if (tid < 64) {
      float a = 1.0f;
      for (int t = 0; t < WKV_L; ++t) { a *= Ac[t * 68 + tid]; Ac[t * 68 + tid] = a; }
      aL[tid] = a;
    }
    __syncthreads();
    // ---- transforms + bf16 state copy ----
    {
#pragma unroll
      for (int q = 0; q < 8; ++q) {
        const int i = li + q;
        const float a  = Ac[lt * 68 + i];
        const float ap = (lt == 0) ? 1.0f : Ac[(lt - 1) * 68 + i];
        const float rt = Rc[lt * 68 + i] * ap;
        const float kt = Kc[lt * 68 + i] / a;
        Rt[lt * 72 + i]   = (__bf16)rt;
        Kt[lt * 72 + i]   = (__bf16)kt;
        BkT[i * 40 + lt]  = (__bf16)(aL[i] * kt);
      }
      const int jr = tid >> 2;            // 0..63
      const int ic = (tid & 3) * 16;
#pragma unroll
      for (int q = 0; q < 16; ++q)
        S0b[jr * 72 + ic + q] = (__bf16)S0T[jr * 68 + ic + q];
    }
    __syncthreads();
    // ---- P = R~ @ K~^T (waves 0-3) ; diagonal (wave 4) ----
    f32x8 pacc;
#pragma unroll
    for (int p = 0; p < 8; ++p) pacc[p] = 0.0f;
    if (wave < 4) {
      const int pmt = wave >> 1, pnt = wave & 1;
#pragma unroll
      for (int kk = 0; kk < 2; ++kk) {
        const bf16x16 af = ld_frag(Rt, pmt * 16, 72, kk * 32, frow, fk);
        const bf16x16 bf_ = ld_frag(Kt, pnt * 16, 72, kk * 32, frow, fk);
        pacc = __builtin_amdgcn_wmma_f32_16x16x32_bf16(
            false, af, false, bf_, (short)0, pacc, false, false);
      }
    } else if (wave == 4) {
      float s = 0.0f;
      for (int i = 0; i < 64; ++i) s += Rc[lane * 68 + i] * su[i] * Kc[lane * 68 + i];
      sdiag[lane] = s;
    }
    __syncthreads();
    // ---- causal mask + u-diagonal; store Pm ----
    if (wave < 4) {
      const int pmt = wave >> 1, pnt = wave & 1;
      const int scol = pnt * 16 + frow;
#pragma unroll
      for (int p = 0; p < 8; ++p) {
        const int trow = pmt * 16 + p + (lane >> 4) * 8;
        const float val = (scol < trow) ? pacc[p]
                        : ((scol == trow) ? sdiag[trow] : 0.0f);
        Pm[trow * 40 + scol] = (__bf16)val;
      }
    }
    __syncthreads();
    // ---- Y = R~ @ S0 + Pm @ V ; write y ----
    {
      const int mt = wave >> 2, nt = wave & 3;
      f32x8 acc;
#pragma unroll
      for (int p = 0; p < 8; ++p) acc[p] = 0.0f;
#pragma unroll
      for (int kk = 0; kk < 2; ++kk) {
        const bf16x16 af = ld_frag(Rt, mt * 16, 72, kk * 32, frow, fk);
        const bf16x16 bf_ = ld_frag(S0b, nt * 16, 72, kk * 32, frow, fk);
        acc = __builtin_amdgcn_wmma_f32_16x16x32_bf16(
            false, af, false, bf_, (short)0, acc, false, false);
      }
      {
        const bf16x16 af = ld_frag(Pm, mt * 16, 40, 0, frow, fk);
        const bf16x16 bf_ = ld_frag(Vt, nt * 16, 40, 0, frow, fk);
        acc = __builtin_amdgcn_wmma_f32_16x16x32_bf16(
            false, af, false, bf_, (short)0, acc, false, false);
      }
      const int j = nt * 16 + frow;
#pragma unroll
      for (int p = 0; p < 8; ++p) {
        const int tl = mt * 16 + p + (lane >> 4) * 8;
        y[base + (size_t)(t0 + tl) * RWKV_C + j] = acc[p];
      }
    }
    // ---- state update: S = diag(aL) S + Bk^T @ V ----
    {
      const int mi = wave >> 1;
#pragma unroll
      for (int q = 0; q < 2; ++q) {
        const int nj = (wave & 1) * 2 + q;
        f32x8 acc;
#pragma unroll
        for (int p = 0; p < 8; ++p) acc[p] = 0.0f;
        const bf16x16 af = ld_frag(BkT, mi * 16, 40, 0, frow, fk);
        const bf16x16 bf_ = ld_frag(Vt, nj * 16, 40, 0, frow, fk);
        acc = __builtin_amdgcn_wmma_f32_16x16x32_bf16(
            false, af, false, bf_, (short)0, acc, false, false);
        const int j  = nj * 16 + frow;
        const int ib = mi * 16 + (lane >> 4) * 8;
        f32x4 o0 = *(f32x4*)&S0T[j * 68 + ib];
        f32x4 o1 = *(f32x4*)&S0T[j * 68 + ib + 4];
#pragma unroll
        for (int p = 0; p < 4; ++p) {
          o0[p] = aL[ib + p] * o0[p] + acc[p];
          o1[p] = aL[ib + 4 + p] * o1[p] + acc[p + 4];
        }
        *(f32x4*)&S0T[j * 68 + ib]     = o0;
        *(f32x4*)&S0T[j * 68 + ib + 4] = o1;
      }
    }
    __syncthreads();
  }
}

// ---------------------------------------------------------------------------
// K6: GroupNorm(H groups, eps=6.4e-4) * g  -> bf16 for output GEMM
// ---------------------------------------------------------------------------
__global__ __launch_bounds__(256) void gn_kernel(const float* __restrict__ y,
                                                 const float* __restrict__ g,
                                                 const float* __restrict__ lnw,
                                                 const float* __restrict__ lnb,
                                                 __bf16* __restrict__ out) {
  const int row = blockIdx.x;
  const int tid = threadIdx.x;
  __shared__ float ssum[256], ssq[256];
  __shared__ float smu[16], srs[16];

  const float* yr = y + (size_t)row * RWKV_C;
  float v4[4];
  float s = 0.0f, sq = 0.0f;
#pragma unroll
  for (int i = 0; i < 4; ++i) {
    const float t = yr[tid * 4 + i];
    v4[i] = t; s += t; sq += t * t;
  }
  ssum[tid] = s; ssq[tid] = sq;
  __syncthreads();
  if (tid < 16) {
    float a = 0.0f, bq = 0.0f;
#pragma unroll
    for (int p = 0; p < 16; ++p) { a += ssum[tid * 16 + p]; bq += ssq[tid * 16 + p]; }
    const float mu  = a * (1.0f / 64.0f);
    const float var = bq * (1.0f / 64.0f) - mu * mu;
    smu[tid] = mu;
    srs[tid] = rsqrtf(var + 6.4e-4f);    // eps = 1e-5 * 8^2
  }
  __syncthreads();
  const int hh = tid >> 4;
  const float mu = smu[hh], rs = srs[hh];
  const float* gr = g + (size_t)row * RWKV_C;
#pragma unroll
  for (int i = 0; i < 4; ++i) {
    const int c = tid * 4 + i;
    const float yn = (v4[i] - mu) * rs * lnw[c] + lnb[c];
    out[(size_t)row * RWKV_C + c] = (__bf16)(yn * gr[c]);
  }
}

// ---------------------------------------------------------------------------
extern "C" void kernel_launch(void* const* d_in, const int* in_sizes, int n_in,
                              void* d_out, int out_size, void* d_ws, size_t ws_size,
                              hipStream_t stream) {
  (void)in_sizes; (void)n_in; (void)out_size; (void)ws_size;
  const float* x    = (const float*)d_in[0];
  const float* tmx  = (const float*)d_in[1];
  const float* tmw  = (const float*)d_in[2];
  const float* tmk  = (const float*)d_in[3];
  const float* tmv  = (const float*)d_in[4];
  const float* tmr  = (const float*)d_in[5];
  const float* tmg  = (const float*)d_in[6];
  const float* maw1 = (const float*)d_in[7];
  const float* maw2 = (const float*)d_in[8];
  const float* tdec = (const float*)d_in[9];
  const float* tdw1 = (const float*)d_in[10];
  const float* tdw2 = (const float*)d_in[11];
  const float* ufaa = (const float*)d_in[12];
  const float* Wr   = (const float*)d_in[13];
  const float* Wk   = (const float*)d_in[14];
  const float* Wv   = (const float*)d_in[15];
  const float* Wg   = (const float*)d_in[16];
  const float* Wo   = (const float*)d_in[17];
  const float* lnw  = (const float*)d_in[18];
  const float* lnb  = (const float*)d_in[19];

  char* ws = (char*)d_ws;
  size_t off = 0;
  auto alloc = [&](size_t bytes) -> void* {
    void* p = ws + off;
    off += (bytes + 255) & ~(size_t)255;
    return p;
  };
  const size_t WB  = (size_t)RWKV_C * RWKV_C * sizeof(__bf16);   // 2 MB
  const size_t AF  = (size_t)RWKV_ROWS * RWKV_C * sizeof(float); // 16 MB
  const size_t AB  = (size_t)RWKV_ROWS * RWKV_C * sizeof(__bf16);// 8 MB

  __bf16* WrT  = (__bf16*)alloc(WB);
  __bf16* WkT  = (__bf16*)alloc(WB);
  __bf16* WvT  = (__bf16*)alloc(WB);
  __bf16* WgT  = (__bf16*)alloc(WB);
  __bf16* WoT  = (__bf16*)alloc(WB);
  float*  t5b  = (float*)alloc((size_t)RWKV_ROWS * 160 * sizeof(float));
  float*  xwb  = (float*)alloc(AF);
  __bf16* xkb  = (__bf16*)alloc(AB);
  __bf16* xvb  = (__bf16*)alloc(AB);
  __bf16* xrb  = (__bf16*)alloc(AB);
  __bf16* xgb  = (__bf16*)alloc(AB);
  float*  rb   = (float*)alloc(AF);
  float*  kb   = (float*)alloc(AF);
  float*  vb   = (float*)alloc(AF);
  float*  gb   = (float*)alloc(AF);
  float*  db   = (float*)alloc(AF);
  float*  yb   = (float*)alloc(AF);
  __bf16* ygn  = (__bf16*)alloc(AB);

  const dim3 tgrid(32, 32);
  wconv_kernel<<<tgrid, 256, 0, stream>>>(Wr, WrT);
  wconv_kernel<<<tgrid, 256, 0, stream>>>(Wk, WkT);
  wconv_kernel<<<tgrid, 256, 0, stream>>>(Wv, WvT);
  wconv_kernel<<<tgrid, 256, 0, stream>>>(Wg, WgT);
  wconv_kernel<<<tgrid, 256, 0, stream>>>(Wo, WoT);

  mix1_kernel<<<RWKV_ROWS, 256, 0, stream>>>(x, tmx, maw1, t5b);
  mix2_kernel<<<RWKV_ROWS, 256, 0, stream>>>(x, tmx, tmw, tmk, tmv, tmr, tmg,
                                             maw2, t5b, xwb, xkb, xvb, xrb, xgb);
  decay_kernel<<<RWKV_ROWS, 256, 0, stream>>>(xwb, tdec, tdw1, tdw2, db);

  const dim3 ggrid(RWKV_C / 128, RWKV_ROWS / 128);
  gemm_bf16_kernel<<<ggrid, 256, 0, stream>>>(xrb, WrT, rb, RWKV_ROWS, RWKV_C, RWKV_C, 0);
  gemm_bf16_kernel<<<ggrid, 256, 0, stream>>>(xkb, WkT, kb, RWKV_ROWS, RWKV_C, RWKV_C, 0);
  gemm_bf16_kernel<<<ggrid, 256, 0, stream>>>(xvb, WvT, vb, RWKV_ROWS, RWKV_C, RWKV_C, 0);
  gemm_bf16_kernel<<<ggrid, 256, 0, stream>>>(xgb, WgT, gb, RWKV_ROWS, RWKV_C, RWKV_C, 1);

  wkv_chunk_kernel<<<32, 256, 0, stream>>>(rb, kb, vb, db, ufaa, yb);
  gn_kernel<<<RWKV_ROWS, 256, 0, stream>>>(yb, gb, lnw, lnb, ygn);

  gemm_bf16_kernel<<<ggrid, 256, 0, stream>>>(ygn, WoT, (float*)d_out,
                                              RWKV_ROWS, RWKV_C, RWKV_C, 0);
}
